// SpectreHead_4088808866353
// MI455X (gfx1250) — matrix operand order repaired and checked by reading, not compiled, not run on hardware
//
#include <hip/hip_runtime.h>
#include <hip/hip_bf16.h>

// ---------------------------------------------------------------- constants
#define BB   4
#define NN   2048
#define DD   1024
#define HH   16
#define DH   64
#define NT   (BB*NN)   // 8192 tokens
#define HID  256
#define GHID 256
#define FQ   1025      // N/2+1 frequency bins
#define FP1  1040      // F padded to multiple of 16 (M-dim of rFFT GEMM)
#define FP2  1056      // F padded to multiple of 32 (K-dim of irFFT GEMM)
#define AAN  32        // anchors
#define KSZ  5
#define LBS  40        // LDS row stride (halves) for a 32-wide K tile: pad 32->40

typedef unsigned short u16t;
typedef __attribute__((ext_vector_type(16))) __bf16 v16bf;
typedef __attribute__((ext_vector_type(8)))  float  v8f;
typedef __attribute__((ext_vector_type(8)))  u16t   u16x8;

union ABFrag { u16x8 h[2]; v16bf v; };

__device__ __forceinline__ u16t f2bf(float f) {
  union { float f; unsigned u; } c; c.f = f;
  unsigned r = c.u + 0x7FFFu + ((c.u >> 16) & 1u);   // round-to-nearest-even
  return (u16t)(r >> 16);
}
__device__ __forceinline__ float gelu_f(float x) {
  return 0.5f * x * (1.f + erff(x * 0.70710678118654752f));
}

// A fragment: 16x32 bf16, row-major source, documented lane layout:
// lane<16: row=lane,     K = k0+0..7  and k0+16..23
// lane>=16: row=lane-16, K = k0+8..15 and k0+24..31
__device__ __forceinline__ v16bf load_a_frag(const u16t* __restrict__ A, int lda,
                                             int r0, int k0, int lane) {
  const u16t* p = A + (r0 + (lane & 15)) * lda + k0 + ((lane >> 4) << 3);
  ABFrag u;
  u.h[0] = *(const u16x8*)(p);
  u.h[1] = *(const u16x8*)(p + 16);
  return u.v;
}
// B fragment (32x16) read from an LDS tile stored as [col][K 0..31] with row
// stride LBS halves. lane<16: col=lane, K=0..7,16..23 ; lane>=16: col=lane-16, K=8..15,24..31
__device__ __forceinline__ v16bf lds_b_frag(const u16t* __restrict__ lds,
                                            int c0loc, int lane) {
  const u16t* p = lds + (c0loc + (lane & 15)) * LBS + ((lane >> 4) << 3);
  ABFrag u;
  u.h[0] = *(const u16x8*)(p);
  u.h[1] = *(const u16x8*)(p + 16);
  return u.v;
}
// Cooperative staging of a 32(K) x 64(col) bf16 tile: 256 threads x 16 bytes.
__device__ __forceinline__ u16x8 stage_b_load(const u16t* __restrict__ Bt, int ldb,
                                              int c0, int k, int tid) {
  int c = tid >> 2, kk = (tid & 3) << 3;
  return *(const u16x8*)(Bt + (c0 + c) * ldb + k + kk);
}
__device__ __forceinline__ void stage_b_store(u16t* __restrict__ lds, int tid,
                                              u16x8 r) {
  int c = tid >> 2, kk = (tid & 3) << 3;
  *(u16x8*)(lds + c * LBS + kk) = r;
}
#define WMMA_BF16(a,b,c) \
  __builtin_amdgcn_wmma_f32_16x16x32_bf16(false,(a),false,(b),(short)0,(c),false,false)

// ---------------------------------------------------------------- prep kernels
__global__ void k_cvt_bf16(const float* __restrict__ in, u16t* __restrict__ out, int n) {
  int i = blockIdx.x * blockDim.x + threadIdx.x;
  if (i < n) out[i] = f2bf(in[i]);
}
// out[c*R + r] = bf16(in[r*C + c])  (transpose + convert)
__global__ void k_tr_bf16(const float* __restrict__ in, u16t* __restrict__ out,
                          int R, int C) {
  int i = blockIdx.x * blockDim.x + threadIdx.x;
  if (i >= R * C) return;
  int r = i / C, c = i - r * C;
  out[c * R + r] = f2bf(in[i]);
}
// Forward DFT bases: Fr[f][n]=cos(2pi f n/N), Fi[f][n]=-sin(2pi f n/N); zero pad rows
__global__ void k_fill_fdft(u16t* __restrict__ Fr, u16t* __restrict__ Fi) {
  int i = blockIdx.x * blockDim.x + threadIdx.x;
  if (i >= FP1 * NN) return;
  int f = i / NN, n = i - f * NN;
  u16t cr = 0, ci = 0;
  if (f < FQ) {
    int ph = (f * n) & (NN - 1);
    float th = (float)ph * (6.283185307179586f / (float)NN);
    cr = f2bf(cosf(th));
    ci = f2bf(-sinf(th));
  }
  Fr[i] = cr; Fi[i] = ci;
}
// Inverse bases with rFFT weights folded in: Ar[n][f]=s_f cos, Ai[n][f]=-s_f sin
__global__ void k_fill_idft(u16t* __restrict__ Ar, u16t* __restrict__ Ai) {
  int i = blockIdx.x * blockDim.x + threadIdx.x;
  if (i >= NN * FP2) return;
  int n = i / FP2, f = i - n * FP2;
  u16t cr = 0, ci = 0;
  if (f < FQ) {
    float sc = (f == 0 || f == NN / 2) ? (1.0f / NN) : (2.0f / NN);
    int ph = (f * n) & (NN - 1);
    float th = (float)ph * (6.283185307179586f / (float)NN);
    cr = f2bf(sc * cosf(th));
    ci = f2bf(-sc * sinf(th));
  }
  Ar[i] = cr; Ai[i] = ci;
}

// ---------------------------------------------------------------- WMMA GEMMs
// v^T = Wv^T @ x^T  (+bv), stored bf16 as v_t[(b*D + m)*N + n]   (B,D,N)
__global__ __launch_bounds__(256) void k_wv_gemm(const u16t* __restrict__ WvT,
    const u16t* __restrict__ xbf, const float* __restrict__ bv,
    u16t* __restrict__ v_t) {
  __shared__ u16t lb[2][64 * LBS];
  const int tid = threadIdx.x, lane = tid & 31, wid = tid >> 5;
  const int m0 = blockIdx.x * 128 + wid * 16;   // output dim rows (per wave)
  const int c0 = blockIdx.y * 64;               // token cols (block uniform)
  v8f acc[4] = {};
  u16x8 sreg = stage_b_load(xbf, DD, c0, 0, tid);
  stage_b_store(lb[0], tid, sreg);
  v16bf a_cur = load_a_frag(WvT, DD, m0, 0, lane);
  __syncthreads();
  for (int k = 0; k < DD; k += 32) {
    const int cur = (k >> 5) & 1;
    const bool more = (k + 32) < DD;
    v16bf a_nxt = a_cur;
    if (more) {                                   // issue next-step loads early
      sreg = stage_b_load(xbf, DD, c0, k + 32, tid);
      a_nxt = load_a_frag(WvT, DD, m0, k + 32, lane);
      __builtin_prefetch((const void*)(WvT + m0 * DD + k + 544), 0, 1);
    }
#pragma unroll
    for (int t = 0; t < 4; ++t)
      acc[t] = WMMA_BF16(a_cur, lds_b_frag(lb[cur], t * 16, lane), acc[t]);
    a_cur = a_nxt;
    if (more) stage_b_store(lb[cur ^ 1], tid, sreg);
    __syncthreads();
  }
  const int lc = lane & 15, lh = (lane >> 4) << 3;
#pragma unroll
  for (int t = 0; t < 4; ++t) {
    int token = c0 + t * 16 + lc;
    int b = token >> 11, n = token & (NN - 1);
#pragma unroll
    for (int g = 0; g < 8; ++g) {
      int m = m0 + g + lh;
      v_t[(b * DD + m) * NN + n] = f2bf(acc[t][g] + bv[m]);
    }
  }
}

// h^T = gelu(Wp1^T @ x^T + bp1), f32, stored h_t[m*NT + token]  (HID, NT)
__global__ __launch_bounds__(256) void k_pool_gemm(const u16t* __restrict__ Wp1T,
    const u16t* __restrict__ xbf, const float* __restrict__ bp1,
    float* __restrict__ h_t) {
  __shared__ u16t lb[2][64 * LBS];
  const int tid = threadIdx.x, lane = tid & 31, wid = tid >> 5;
  const int m0 = blockIdx.x * 128 + wid * 16;
  const int c0 = blockIdx.y * 64;
  v8f acc[4] = {};
  u16x8 sreg = stage_b_load(xbf, DD, c0, 0, tid);
  stage_b_store(lb[0], tid, sreg);
  v16bf a_cur = load_a_frag(Wp1T, DD, m0, 0, lane);
  __syncthreads();
  for (int k = 0; k < DD; k += 32) {
    const int cur = (k >> 5) & 1;
    const bool more = (k + 32) < DD;
    v16bf a_nxt = a_cur;
    if (more) {
      sreg = stage_b_load(xbf, DD, c0, k + 32, tid);
      a_nxt = load_a_frag(Wp1T, DD, m0, k + 32, lane);
    }
#pragma unroll
    for (int t = 0; t < 4; ++t)
      acc[t] = WMMA_BF16(a_cur, lds_b_frag(lb[cur], t * 16, lane), acc[t]);
    a_cur = a_nxt;
    if (more) stage_b_store(lb[cur ^ 1], tid, sreg);
    __syncthreads();
  }
  const int lc = lane & 15, lh = (lane >> 4) << 3;
#pragma unroll
  for (int t = 0; t < 4; ++t) {
    int token = c0 + t * 16 + lc;
#pragma unroll
    for (int g = 0; g < 8; ++g) {
      int m = m0 + g + lh;
      h_t[m * NT + token] = gelu_f(acc[t][g] + bp1[m]);
    }
  }
}

// rFFT as GEMM + fused complex spectral gate.
// Vr = Fr @ v, Vi = Fi @ v ; Z = (Vr + i Vi) * (gr + i gi) -> bf16 (B,H,dh,FP2)
__global__ __launch_bounds__(256) void k_rfft_gemm(const u16t* __restrict__ Fr,
    const u16t* __restrict__ Fi, const u16t* __restrict__ v_t,
    const float* __restrict__ gr, const float* __restrict__ gi,
    u16t* __restrict__ Zr, u16t* __restrict__ Zi) {
  __shared__ u16t lb[2][64 * LBS];
  const int tid = threadIdx.x, lane = tid & 31, wid = tid >> 5;
  const int tfRaw = blockIdx.x * 8 + wid;
  const int tf = min(tfRaw, FP1 / 16 - 1);      // clamp: keep barriers + EXEC=~0
  const int f0 = tf * 16;
  const int bh = blockIdx.z;
  const u16t* Bt = v_t + (bh * DH) * NN;        // rows = dh channels, K = n
  v8f accR[4] = {}, accI[4] = {};
  u16x8 sreg = stage_b_load(Bt, NN, 0, 0, tid);
  stage_b_store(lb[0], tid, sreg);
  v16bf aR = load_a_frag(Fr, NN, f0, 0, lane);
  v16bf aI = load_a_frag(Fi, NN, f0, 0, lane);
  __syncthreads();
  for (int k = 0; k < NN; k += 32) {
    const int cur = (k >> 5) & 1;
    const bool more = (k + 32) < NN;
    v16bf aRn = aR, aIn = aI;
    if (more) {
      sreg = stage_b_load(Bt, NN, 0, k + 32, tid);
      aRn = load_a_frag(Fr, NN, f0, k + 32, lane);
      aIn = load_a_frag(Fi, NN, f0, k + 32, lane);
    }
#pragma unroll
    for (int t = 0; t < 4; ++t) {
      v16bf b = lds_b_frag(lb[cur], t * 16, lane);   // reused for cos & sin
      accR[t] = WMMA_BF16(aR, b, accR[t]);
      accI[t] = WMMA_BF16(aI, b, accI[t]);
    }
    aR = aRn; aI = aIn;
    if (more) stage_b_store(lb[cur ^ 1], tid, sreg);
    __syncthreads();
  }
  if (tfRaw >= FP1 / 16) return;                // drop redundant clamped tiles
  const int lc = lane & 15, lh = (lane >> 4) << 3;
#pragma unroll
  for (int t = 0; t < 4; ++t) {
#pragma unroll
    for (int g = 0; g < 8; ++g) {
      int f = f0 + g + lh;
      float g_r = gr[bh * FP1 + f], g_i = gi[bh * FP1 + f];
      float vr = accR[t][g], vi = accI[t][g];
      int zo = (bh * DH + t * 16 + lc) * FP2 + f;
      Zr[zo] = f2bf(vr * g_r - vi * g_i);
      Zi[zo] = f2bf(vr * g_i + vi * g_r);
    }
  }
}

// irFFT as GEMM: y = Ar @ Zr + Ai @ Zi -> bf16 y[(b*N + n)*D + h*dh + c]
__global__ __launch_bounds__(256) void k_irfft_gemm(const u16t* __restrict__ Ar,
    const u16t* __restrict__ Ai, const u16t* __restrict__ Zr,
    const u16t* __restrict__ Zi, u16t* __restrict__ ybf) {
  __shared__ u16t lbR[2][64 * LBS];
  __shared__ u16t lbI[2][64 * LBS];
  const int tid = threadIdx.x, lane = tid & 31, wid = tid >> 5;
  const int n0 = (blockIdx.x * 8 + wid) * 16;
  const int bh = blockIdx.z;
  const int b = bh >> 4, h = bh & 15;
  const u16t* BtR = Zr + (bh * DH) * FP2;
  const u16t* BtI = Zi + (bh * DH) * FP2;
  v8f acc[4] = {};
  u16x8 srR = stage_b_load(BtR, FP2, 0, 0, tid);
  u16x8 srI = stage_b_load(BtI, FP2, 0, 0, tid);
  stage_b_store(lbR[0], tid, srR);
  stage_b_store(lbI[0], tid, srI);
  v16bf aR = load_a_frag(Ar, FP2, n0, 0, lane);
  v16bf aI = load_a_frag(Ai, FP2, n0, 0, lane);
  __syncthreads();
  for (int k = 0; k < FP2; k += 32) {
    const int cur = (k >> 5) & 1;
    const bool more = (k + 32) < FP2;
    v16bf aRn = aR, aIn = aI;
    if (more) {
      srR = stage_b_load(BtR, FP2, 0, k + 32, tid);
      srI = stage_b_load(BtI, FP2, 0, k + 32, tid);
      aRn = load_a_frag(Ar, FP2, n0, k + 32, lane);
      aIn = load_a_frag(Ai, FP2, n0, k + 32, lane);
    }
#pragma unroll
    for (int t = 0; t < 4; ++t) {
      acc[t] = WMMA_BF16(aR, lds_b_frag(lbR[cur], t * 16, lane), acc[t]);
      acc[t] = WMMA_BF16(aI, lds_b_frag(lbI[cur], t * 16, lane), acc[t]);
    }
    aR = aRn; aI = aIn;
    if (more) {
      stage_b_store(lbR[cur ^ 1], tid, srR);
      stage_b_store(lbI[cur ^ 1], tid, srI);
    }
    __syncthreads();
  }
  const int lc = lane & 15, lh = (lane >> 4) << 3;
#pragma unroll
  for (int t = 0; t < 4; ++t) {
    int c = t * 16 + lc;
#pragma unroll
    for (int g = 0; g < 8; ++g) {
      int n = n0 + g + lh;
      ybf[(b * NN + n) * DD + h * DH + c] = f2bf(acc[t][g]);
    }
  }
}

// out = y @ Wo + bo  (A = y rows=tokens, Bt = Wo^T rows=out-dim), f32 output
__global__ __launch_bounds__(256) void k_out_gemm(const u16t* __restrict__ ybf,
    const u16t* __restrict__ WoT, const float* __restrict__ bo,
    float* __restrict__ out) {
  __shared__ u16t lb[2][64 * LBS];
  const int tid = threadIdx.x, lane = tid & 31, wid = tid >> 5;
  const int r0 = blockIdx.x * 128 + wid * 16;   // token rows
  const int c0 = blockIdx.y * 64;               // out-dim cols
  v8f acc[4] = {};
  u16x8 sreg = stage_b_load(WoT, DD, c0, 0, tid);
  stage_b_store(lb[0], tid, sreg);
  v16bf a_cur = load_a_frag(ybf, DD, r0, 0, lane);
  __syncthreads();
  for (int k = 0; k < DD; k += 32) {
    const int cur = (k >> 5) & 1;
    const bool more = (k + 32) < DD;
    v16bf a_nxt = a_cur;
    if (more) {
      sreg = stage_b_load(WoT, DD, c0, k + 32, tid);
      a_nxt = load_a_frag(ybf, DD, r0, k + 32, lane);
      __builtin_prefetch((const void*)(ybf + r0 * DD + k + 544), 0, 1);
    }
#pragma unroll
    for (int t = 0; t < 4; ++t)
      acc[t] = WMMA_BF16(a_cur, lds_b_frag(lb[cur], t * 16, lane), acc[t]);
    a_cur = a_nxt;
    if (more) stage_b_store(lb[cur ^ 1], tid, sreg);
    __syncthreads();
  }
  const int lc = lane & 15, lh = (lane >> 4) << 3;
#pragma unroll
  for (int t = 0; t < 4; ++t) {
    int dim = c0 + t * 16 + lc;
    float bias = bo[dim];
#pragma unroll
    for (int g = 0; g < 8; ++g) {
      int token = r0 + g + lh;
      out[token * DD + dim] = acc[t][g] + bias;
    }
  }
}

// ---------------------------------------------------------------- small kernels
__global__ void k_score(const float* __restrict__ h_t, const float* __restrict__ Wp2,
                        const float* __restrict__ bp2, float* __restrict__ s) {
  int t = blockIdx.x * blockDim.x + threadIdx.x;
  if (t >= NT) return;
  float acc = bp2[0];
  for (int d = 0; d < HID; ++d) acc += h_t[d * NT + t] * Wp2[d];
  s[t] = acc;
}

__global__ void k_softmax(const float* __restrict__ s, float* __restrict__ w) {
  __shared__ float red[256];
  int b = blockIdx.x, tid = threadIdx.x;
  const float* sb = s + b * NN;
  float m = -1e30f;
  for (int n = tid; n < NN; n += 256) m = fmaxf(m, sb[n]);
  red[tid] = m; __syncthreads();
  for (int st = 128; st; st >>= 1) {
    if (tid < st) red[tid] = fmaxf(red[tid], red[tid + st]);
    __syncthreads();
  }
  m = red[0]; __syncthreads();
  float sum = 0.f;
  for (int n = tid; n < NN; n += 256) sum += __expf(sb[n] - m);
  red[tid] = sum; __syncthreads();
  for (int st = 128; st; st >>= 1) {
    if (tid < st) red[tid] += red[tid + st];
    __syncthreads();
  }
  float inv = 1.f / red[0];
  for (int n = tid; n < NN; n += 256) w[b * NN + n] = __expf(sb[n] - m) * inv;
}

__global__ void k_wsum(const float* __restrict__ w, const float* __restrict__ x,
                       float* __restrict__ gv) {
  int b = blockIdx.x;
  int d = blockIdx.y * 256 + threadIdx.x;
  float acc = 0.f;
  for (int n = 0; n < NN; ++n) acc += w[b * NN + n] * x[(b * NN + n) * DD + d];
  gv[b * DD + d] = acc;
}

__global__ void k_gate_mlp1(const float* __restrict__ gv, const float* __restrict__ Wg1,
                            const float* __restrict__ bg1, float* __restrict__ gh) {
  int i = blockIdx.x * blockDim.x + threadIdx.x;
  if (i >= BB * GHID) return;
  int b = i >> 8, j = i & 255;
  float acc = bg1[j];
  for (int d = 0; d < DD; ++d) acc += gv[b * DD + d] * Wg1[d * GHID + j];
  gh[i] = gelu_f(acc);
}

__global__ void k_gate_mlp2(const float* __restrict__ gh, const float* __restrict__ Wg2,
                            const float* __restrict__ bg2, float* __restrict__ an) {
  int i = blockIdx.x * blockDim.x + threadIdx.x;
  if (i >= BB * 2 * HH * AAN) return;
  int b = i >> 10, o = i & 1023;
  float acc = bg2[o];
  for (int j = 0; j < GHID; ++j) acc += gh[b * GHID + j] * Wg2[j * 1024 + o];
  an[i] = acc;
}

// cubic interp (Keys a=-0.5) of 32 anchors -> F bins, modReLU, circular conv5
__global__ void k_gate(const float* __restrict__ an, const float* __restrict__ mb,
                       const float* __restrict__ ckr, const float* __restrict__ cki,
                       float* __restrict__ gr, float* __restrict__ gi) {
  __shared__ float sr[FQ], si[FQ];
  const int bh = blockIdx.x;
  const float* ap = an + bh * AAN * 2;
  for (int f = threadIdx.x; f < FQ; f += blockDim.x) {
    float xx = (f + 0.5f) * ((float)AAN / (float)FQ) - 0.5f;
    int base = (int)floorf(xx);
    float wr = 0.f, wi = 0.f, ws = 0.f;
#pragma unroll
    for (int j = -1; j <= 2; ++j) {
      float t = fabsf(xx - (float)(base + j));
      float wt;
      if (t <= 1.f)      wt = ((1.5f * t - 2.5f) * t) * t + 1.f;
      else if (t < 2.f)  wt = -0.5f * (((t - 5.f) * t + 8.f) * t - 4.f);
      else               wt = 0.f;
      int idx = min(max(base + j, 0), AAN - 1);
      wr += wt * ap[idx * 2 + 0];
      wi += wt * ap[idx * 2 + 1];
      ws += wt;
    }
    wr /= ws; wi /= ws;
    float mag = sqrtf(wr * wr + wi * wi);
    float sc = fmaxf(mag + mb[f], 0.f) / (mag + 1e-8f);
    sr[f] = wr * sc; si[f] = wi * sc;
  }
  __syncthreads();
  float kr[KSZ], ki[KSZ];
#pragma unroll
  for (int j = 0; j < KSZ; ++j) { kr[j] = ckr[j]; ki[j] = cki[j]; }
  for (int f = threadIdx.x; f < FP1; f += blockDim.x) {
    float orr = 0.f, oii = 0.f;
    if (f < FQ) {
#pragma unroll
      for (int j = 0; j < KSZ; ++j) {
        int idx = (f + j - 2 + FQ) % FQ;
        orr += kr[j] * sr[idx] - ki[j] * si[idx];
        oii += kr[j] * si[idx] + ki[j] * sr[idx];
      }
    }
    gr[bh * FP1 + f] = orr;   // zero in padded rows -> padded spectra are zero
    gi[bh * FP1 + f] = oii;
  }
}

// ---------------------------------------------------------------- launch
extern "C" void kernel_launch(void* const* d_in, const int* in_sizes, int n_in,
                              void* d_out, int out_size, void* d_ws, size_t ws_size,
                              hipStream_t stream) {
  const float* x   = (const float*)d_in[0];
  const float* Wv  = (const float*)d_in[1];
  const float* bv  = (const float*)d_in[2];
  const float* Wp1 = (const float*)d_in[3];
  const float* bp1 = (const float*)d_in[4];
  const float* Wp2 = (const float*)d_in[5];
  const float* bp2 = (const float*)d_in[6];
  const float* Wg1 = (const float*)d_in[7];
  const float* bg1 = (const float*)d_in[8];
  const float* Wg2 = (const float*)d_in[9];
  const float* bg2 = (const float*)d_in[10];
  const float* mb  = (const float*)d_in[11];
  const float* ckr = (const float*)d_in[12];
  const float* cki = (const float*)d_in[13];
  const float* Wo  = (const float*)d_in[14];
  const float* bo  = (const float*)d_in[15];
  float* out = (float*)d_out;

  char* p = (char*)d_ws;
  auto alloc = [&](size_t bytes) -> char* {
    char* r = p; p += (bytes + 255) & ~(size_t)255; return r;
  };
  u16t*  xbf  = (u16t*)alloc((size_t)NT * DD * 2);
  u16t*  WvT  = (u16t*)alloc((size_t)DD * DD * 2);
  u16t*  Wp1T = (u16t*)alloc((size_t)HID * DD * 2);
  u16t*  WoT  = (u16t*)alloc((size_t)DD * DD * 2);
  u16t*  v_t  = (u16t*)alloc((size_t)BB * DD * NN * 2);
  float* h_t  = (float*)alloc((size_t)HID * NT * 4);
  float* s    = (float*)alloc((size_t)NT * 4);
  float* w    = (float*)alloc((size_t)NT * 4);
  float* gv   = (float*)alloc((size_t)BB * DD * 4);
  float* ghid = (float*)alloc((size_t)BB * GHID * 4);
  float* an   = (float*)alloc((size_t)BB * 2 * HH * AAN * 4);
  float* gr   = (float*)alloc((size_t)BB * HH * FP1 * 4);
  float* gi   = (float*)alloc((size_t)BB * HH * FP1 * 4);
  u16t*  Fr   = (u16t*)alloc((size_t)FP1 * NN * 2);
  u16t*  Fi   = (u16t*)alloc((size_t)FP1 * NN * 2);
  u16t*  Ar   = (u16t*)alloc((size_t)NN * FP2 * 2);
  u16t*  Ai   = (u16t*)alloc((size_t)NN * FP2 * 2);
  u16t*  Zr   = (u16t*)alloc((size_t)BB * HH * DH * FP2 * 2);
  u16t*  Zi   = (u16t*)alloc((size_t)BB * HH * DH * FP2 * 2);
  u16t*  ybf  = (u16t*)alloc((size_t)NT * DD * 2);
  (void)ws_size; (void)n_in; (void)in_sizes; (void)out_size;

  // prep: bf16 activations, transposed bf16 weights, DFT bases
  k_cvt_bf16<<<(NT * DD) / 256, 256, 0, stream>>>(x, xbf, NT * DD);
  k_tr_bf16 <<<(DD * DD) / 256, 256, 0, stream>>>(Wv, WvT, DD, DD);
  k_tr_bf16 <<<(DD * HID) / 256, 256, 0, stream>>>(Wp1, Wp1T, DD, HID);
  k_tr_bf16 <<<(DD * DD) / 256, 256, 0, stream>>>(Wo, WoT, DD, DD);
  k_fill_fdft<<<(FP1 * NN) / 256, 256, 0, stream>>>(Fr, Fi);
  k_fill_idft<<<(NN * FP2) / 256, 256, 0, stream>>>(Ar, Ai);

  // big WMMA GEMMs + pooling path
  k_wv_gemm  <<<dim3(DD / 128, NT / 64), 256, 0, stream>>>(WvT, xbf, bv, v_t);
  k_pool_gemm<<<dim3(HID / 128, NT / 64), 256, 0, stream>>>(Wp1T, xbf, bp1, h_t);
  k_score    <<<NT / 256, 256, 0, stream>>>(h_t, Wp2, bp2, s);
  k_softmax  <<<BB, 256, 0, stream>>>(s, w);
  k_wsum     <<<dim3(BB, DD / 256), 256, 0, stream>>>(w, x, gv);
  k_gate_mlp1<<<(BB * GHID) / 256, 256, 0, stream>>>(gv, Wg1, bg1, ghid);
  k_gate_mlp2<<<(BB * 2 * HH * AAN) / 256, 256, 0, stream>>>(ghid, Wg2, bg2, an);
  k_gate     <<<BB * HH, 256, 0, stream>>>(an, mb, ckr, cki, gr, gi);

  // spectral path as WMMA DFT GEMMs with fused gate, then output projection
  k_rfft_gemm <<<dim3((FP1 / 16 + 7) / 8, 1, BB * HH), 256, 0, stream>>>(
      Fr, Fi, v_t, gr, gi, Zr, Zi);
  k_irfft_gemm<<<dim3(NN / 128, 1, BB * HH), 256, 0, stream>>>(Ar, Ai, Zr, Zi, ybf);
  k_out_gemm  <<<dim3(NT / 128, DD / 64), 256, 0, stream>>>(ybf, WoT, bo, out);
}
